// Attention_19739669692939
// MI455X (gfx1250) — matrix operand order repaired and checked by reading, not compiled, not run on hardware
//
#include <hip/hip_runtime.h>
#include <hip/hip_bf16.h>

#define L_  1024
#define B_  64
#define C_  1024   // ENC2
#define D_  512    // DEC
#define E_  512
#define WCOLS (C_ + D_)   // 1536, row stride of W_attn

typedef __attribute__((ext_vector_type(16))) _Float16 v16h;
typedef __attribute__((ext_vector_type(8)))  _Float16 v8h;
typedef __attribute__((ext_vector_type(4)))  _Float16 v4h;
typedef __attribute__((ext_vector_type(8)))  float    v8f;
typedef __attribute__((ext_vector_type(4)))  float    v4f;

// ---------------------------------------------------------------------------
// Kernel 1: sbias[b][e] = sum_d s[b,d] * W_attn[e, 1024 + d]
// ---------------------------------------------------------------------------
__global__ void bias_kernel(const float* __restrict__ s,
                            const float* __restrict__ W,
                            float* __restrict__ sbias) {
    const int b = blockIdx.x;      // 0..63
    const int e = threadIdx.x;     // 0..511
    const float* sr = s + b * D_;
    const float* wr = W + (size_t)e * WCOLS + C_;
    float acc = 0.f;
    #pragma unroll 4
    for (int d = 0; d < D_; d += 4) {
        v4f sv = *(const v4f*)(sr + d);
        v4f wv = *(const v4f*)(wr + d);
        acc += sv.x * wv.x + sv.y * wv.y + sv.z * wv.z + sv.w * wv.w;
    }
    sbias[b * E_ + e] = acc;
}

// ---------------------------------------------------------------------------
// Kernel 2: fused GEMM(f16 WMMA, f32 acc) + bias + tanh + dot(W_v) reduction.
// M-tile = 128 rows/WG (2 'l' values x 64 batches) to halve Wa_e L2 traffic.
// 32 waves; wave w owns N-tile n0 = w*16 and 8 M-subtiles of 16.
// A chunks stream into double-buffered LDS via GLOBAL_LOAD_ASYNC_TO_LDS_B128
// (ASYNCcnt), overlapping DMA with WMMA.
// ---------------------------------------------------------------------------
#define MT      128               // rows per workgroup
#define KC      64                // K-chunk staged in LDS
#define NCHUNK  (C_ / KC)         // 16
#define ROWPADF 68                // f32 elements per LDS A row (64 + 4 pad)

__global__ __launch_bounds__(1024)
void attn_gemm_kernel(const float* __restrict__ enc,
                      const float* __restrict__ W,
                      const float* __restrict__ sbias,
                      const float* __restrict__ Wv,
                      float* __restrict__ logits) {
    __shared__ float sAf[2][MT * ROWPADF];   // 2 x 34816 B
    __shared__ float s_logit[MT];

    const int tid    = threadIdx.x;
    const int wave   = tid >> 5;          // 0..31 -> N tile
    const int lane   = tid & 31;
    const int laneLo = lane & 15;
    const int half   = lane >> 4;         // 0 or 1
    const int m0     = blockIdx.x * MT;   // global A row base
    const int n0     = wave * 16;

    if (tid < MT) s_logit[tid] = 0.f;

    v8f acc[8] = {};

    // ---- async staging: each thread copies 32 B (2 x b128) per chunk ----
    const int sr = tid >> 3;              // row 0..127
    const int sc = (tid & 7) * 8;         // col 0..56 step 8
    // LDS byte addresses (wave-relative = low 32 bits of flat pointer)
    const unsigned lds0 = (unsigned)(size_t)&sAf[0][sr * ROWPADF + sc];
    const unsigned lds1 = (unsigned)(size_t)&sAf[1][sr * ROWPADF + sc];
    // byte offset of this thread's element from enc base (fits in u32: < 2^29)
    const unsigned gthread = (unsigned)(((m0 + sr) * C_ + sc) * 4);

    // issue async DMA of chunk k0 into LDS buffer at ldsaddr (2 x 16B)
    auto issue = [&](int k0, unsigned ldsaddr) {
        unsigned go = gthread + (unsigned)(k0 * 4);
        asm volatile("global_load_async_to_lds_b128 %0, %1, %2"
                     :: "v"(ldsaddr), "v"(go), "s"(enc) : "memory");
        asm volatile("global_load_async_to_lds_b128 %0, %1, %2"
                     :: "v"(ldsaddr + 16u), "v"(go + 16u), "s"(enc) : "memory");
    };

    // B column pointer: lane holds column n = n0 + laneLo of B (= row n of Wa_e)
    const float* wcol = W + (size_t)(n0 + laneLo) * WCOLS;

    issue(0, lds0);

    for (int ch = 0; ch < NCHUNK; ++ch) {
        const int k0 = ch * KC;
        if (ch + 1 < NCHUNK) {
            issue(k0 + KC, (ch & 1) ? lds0 : lds1);     // prefetch next chunk
            asm volatile("s_wait_asynccnt 0x2" ::: "memory"); // current 2 landed
        } else {
            asm volatile("s_wait_asynccnt 0x0" ::: "memory");
        }
        __syncthreads();

        const float* Abuf = sAf[ch & 1];

        #pragma unroll
        for (int ks = 0; ks < KC; ks += 32) {
            // ---- B fragment: lane n=n0+laneLo holds K = (k0+ks) + {0..7,16..23} (half 0) / {8..15,24..31} (half 1)
            const float* bp = wcol + k0 + ks + half * 8;
            v4f b0 = *(const v4f*)(bp + 0);
            v4f b1 = *(const v4f*)(bp + 4);
            v4f b2 = *(const v4f*)(bp + 16);
            v4f b3 = *(const v4f*)(bp + 20);
            v16h bf;
            bf[0]  = (_Float16)b0.x; bf[1]  = (_Float16)b0.y;
            bf[2]  = (_Float16)b0.z; bf[3]  = (_Float16)b0.w;
            bf[4]  = (_Float16)b1.x; bf[5]  = (_Float16)b1.y;
            bf[6]  = (_Float16)b1.z; bf[7]  = (_Float16)b1.w;
            bf[8]  = (_Float16)b2.x; bf[9]  = (_Float16)b2.y;
            bf[10] = (_Float16)b2.z; bf[11] = (_Float16)b2.w;
            bf[12] = (_Float16)b3.x; bf[13] = (_Float16)b3.y;
            bf[14] = (_Float16)b3.z; bf[15] = (_Float16)b3.w;

            // ---- A fragments in 2 groups of 4 (bounds register pressure,
            //      still lets the 16 ds_load_b128 of a group pipeline)
            // lane row = mt*16+laneLo; K offs {0..7,16..23} (half 0) / {8..15,24..31}
            #pragma unroll
            for (int g = 0; g < 2; ++g) {
                v16h af[4];
                #pragma unroll
                for (int q = 0; q < 4; ++q) {
                    const int mt = g * 4 + q;
                    const float* ap = &Abuf[(mt * 16 + laneLo) * ROWPADF + ks + half * 8];
                    v4f x0 = *(const v4f*)(ap + 0);
                    v4f x1 = *(const v4f*)(ap + 4);
                    v4f x2 = *(const v4f*)(ap + 16);
                    v4f x3 = *(const v4f*)(ap + 20);
                    v16h t;
                    t[0]  = (_Float16)x0.x; t[1]  = (_Float16)x0.y;
                    t[2]  = (_Float16)x0.z; t[3]  = (_Float16)x0.w;
                    t[4]  = (_Float16)x1.x; t[5]  = (_Float16)x1.y;
                    t[6]  = (_Float16)x1.z; t[7]  = (_Float16)x1.w;
                    t[8]  = (_Float16)x2.x; t[9]  = (_Float16)x2.y;
                    t[10] = (_Float16)x2.z; t[11] = (_Float16)x2.w;
                    t[12] = (_Float16)x3.x; t[13] = (_Float16)x3.y;
                    t[14] = (_Float16)x3.z; t[15] = (_Float16)x3.w;
                    af[q] = t;
                }
                #pragma unroll
                for (int q = 0; q < 4; ++q) {
                    acc[g * 4 + q] = __builtin_amdgcn_wmma_f32_16x16x32_f16(
                                         false, af[q], false, bf, (short)0,
                                         acc[g * 4 + q], false, false);
                }
            }
        }
        __syncthreads();   // all waves done with this buffer before it is re-DMAed
    }

    // ---- epilogue: tanh(acc + sbias) dotted with W_v, reduced over e ----
    const float wv_n = Wv[n0 + laneLo];
    #pragma unroll
    for (int mt = 0; mt < 8; ++mt) {
        #pragma unroll
        for (int v = 0; v < 8; ++v) {
            const int r = mt * 16 + half * 8 + v;   // 0..127; batch b = r & 63
            float pre = acc[mt][v] + sbias[(r & 63) * E_ + n0 + laneLo];
            float p = wv_n * tanhf(pre);
            p += __shfl_xor(p, 1);
            p += __shfl_xor(p, 2);
            p += __shfl_xor(p, 4);
            p += __shfl_xor(p, 8);
            if (laneLo == 0) atomicAdd(&s_logit[r], p);
        }
    }
    __syncthreads();

    if (tid < MT) {
        const int b = tid & 63;
        const int l = blockIdx.x * 2 + (tid >> 6);
        logits[(size_t)b * L_ + l] = s_logit[tid];
    }
}

// ---------------------------------------------------------------------------
// Kernel 3: row softmax over L per batch. logits: (B, L) -> out: (B, L)
// ---------------------------------------------------------------------------
__global__ void softmax_kernel(const float* __restrict__ logits,
                               float* __restrict__ out) {
    __shared__ float red[8];
    __shared__ float bcast;
    const int b   = blockIdx.x;
    const int tid = threadIdx.x;   // 256 threads = 8 waves
    const float* row = logits + (size_t)b * L_;

    float m = -3.402823466e38f;
    for (int i = tid; i < L_; i += 256) m = fmaxf(m, row[i]);
    #pragma unroll
    for (int o = 16; o > 0; o >>= 1) m = fmaxf(m, __shfl_xor(m, o));
    if ((tid & 31) == 0) red[tid >> 5] = m;
    __syncthreads();
    if (tid == 0) {
        float mm = red[0];
        #pragma unroll
        for (int i = 1; i < 8; ++i) mm = fmaxf(mm, red[i]);
        bcast = mm;
    }
    __syncthreads();
    const float M = bcast;

    float s = 0.f;
    for (int i = tid; i < L_; i += 256) s += __expf(row[i] - M);
    #pragma unroll
    for (int o = 16; o > 0; o >>= 1) s += __shfl_xor(s, o);
    if ((tid & 31) == 0) red[tid >> 5] = s;
    __syncthreads();
    if (tid == 0) {
        float ss = 0.f;
        #pragma unroll
        for (int i = 0; i < 8; ++i) ss += red[i];
        bcast = ss;
    }
    __syncthreads();
    const float inv = 1.f / bcast;

    for (int i = tid; i < L_; i += 256)
        out[(size_t)b * L_ + i] = __expf(row[i] - M) * inv;
}

// ---------------------------------------------------------------------------
extern "C" void kernel_launch(void* const* d_in, const int* in_sizes, int n_in,
                              void* d_out, int out_size, void* d_ws, size_t ws_size,
                              hipStream_t stream) {
    const float* enc = (const float*)d_in[0];   // (1024, 64, 1024)
    const float* s   = (const float*)d_in[1];   // (1, 64, 512)
    const float* W   = (const float*)d_in[2];   // (512, 1536)
    const float* Wv  = (const float*)d_in[3];   // (1, 512)
    float* out = (float*)d_out;                 // (64, 1024)

    float* sbias  = (float*)d_ws;               // 64*512 f32
    float* logits = sbias + B_ * E_;            // 64*1024 f32

    bias_kernel<<<B_, E_, 0, stream>>>(s, W, sbias);
    attn_gemm_kernel<<<L_ / 2, 1024, 0, stream>>>(enc, W, sbias, Wv, logits);
    softmax_kernel<<<B_, 256, 0, stream>>>(logits, out);
}
